// Encoder_79680233275453
// MI455X (gfx1250) — matrix operand order repaired and checked by reading, not compiled, not run on hardware
//
#include <hip/hip_runtime.h>
#include <hip/hip_bf16.h>

typedef __attribute__((ext_vector_type(16))) _Float16 v16h;
typedef __attribute__((ext_vector_type(8)))  _Float16 v8h;
typedef __attribute__((ext_vector_type(2)))  _Float16 v2h;
typedef __attribute__((ext_vector_type(8)))  float    v8f;

#define NB    8
#define NPTS  4096
#define KNN   20
#define CH    64
#define TST   40          // transposed-tile row stride (halves); 80B rows, 16B-aligned subloads
#define BNKF  655360.0f   // 8*4096*20
#define EPSV  1e-5f

#define SHUF16(lo, hi) __builtin_shufflevector((lo), (hi), 0,1,2,3,4,5,6,7,8,9,10,11,12,13,14,15)

// ---------------------------------------------------------------------------
// Kernel 1: brute-force KNN (top-20 smallest distances per row).
// One block per (batch, 256 rows); batch points live in LDS as float4
// (x,y,z,|p|^2) so the scan loop is one ds_load_b128 per candidate.
// ---------------------------------------------------------------------------
__global__ __launch_bounds__(256) void knn_kernel(const float* __restrict__ x,
                                                  int* __restrict__ idx_out) {
  __shared__ float4 sp[NPTS];
  const int b = blockIdx.y;
  const float* xb = x + (size_t)b * 3 * NPTS;
  for (int i = threadIdx.x; i < NPTS; i += 256) {
    float px = xb[i], py = xb[NPTS + i], pz = xb[2 * NPTS + i];
    sp[i] = make_float4(px, py, pz, px * px + py * py + pz * pz);
  }
  __syncthreads();

  const int n = blockIdx.x * 256 + threadIdx.x;
  const float4 p = sp[n];
  const float px = p.x, py = p.y, pz = p.z;

  float best[KNN]; int bidx[KNN];
  float wmax = -1e30f; int wpos = 0;
#pragma unroll
  for (int j = 0; j < KNN; j++) {
    float4 q = sp[j];
    float dot = px * q.x + py * q.y + pz * q.z;
    best[j] = fmaf(-2.f, dot, q.w);
    bidx[j] = j;
  }
#pragma unroll
  for (int j = 0; j < KNN; j++) {
    if (best[j] > wmax) { wmax = best[j]; wpos = j; }
  }
  for (int m = KNN; m < NPTS; m++) {
    float4 q = sp[m];
    float dot = px * q.x + py * q.y + pz * q.z;
    float e = fmaf(-2.f, dot, q.w);
    if (e < wmax) {
#pragma unroll
      for (int j = 0; j < KNN; j++)
        if (j == wpos) { best[j] = e; bidx[j] = m; }
      wmax = -1e30f;
#pragma unroll
      for (int j = 0; j < KNN; j++)
        if (best[j] > wmax) { wmax = best[j]; wpos = j; }
    }
  }
  // rank-based write: ascending (value, index), matching top_k(-dist) order
  int* op = idx_out + ((size_t)b * NPTS + n) * KNN;
#pragma unroll
  for (int j = 0; j < KNN; j++) {
    int r = 0;
#pragma unroll
    for (int i = 0; i < KNN; i++) {
      bool lt = (best[i] < best[j]) ||
                (best[i] == best[j] && bidx[i] < bidx[j]);
      r += (i != j && lt) ? 1 : 0;
    }
    op[r] = bidx[j];
  }
}

// ---------------------------------------------------------------------------
// Kernel 2: first/second moments of the 6-dim edge feature over (B,N,K).
// Per-block partial sums (fixed order -> deterministic).
// ---------------------------------------------------------------------------
__global__ __launch_bounds__(256) void moments_kernel(const float* __restrict__ x,
                                                      const int* __restrict__ idx,
                                                      float* __restrict__ part) {
  const int t = blockIdx.x * 256 + threadIdx.x;       // point id 0..32767
  const int b = t >> 12, n = t & (NPTS - 1);
  const float* xb = x + (size_t)b * 3 * NPTS;
  const float px = xb[n], py = xb[NPTS + n], pz = xb[2 * NPTS + n];
  const int* ip = idx + (size_t)t * KNN;

  float loc[27];
#pragma unroll
  for (int q = 0; q < 27; q++) loc[q] = 0.f;

#pragma unroll 1
  for (int k = 0; k < KNN; k++) {
    const int j = ip[k];
    float e[6];
    e[0] = xb[j] - px; e[1] = xb[NPTS + j] - py; e[2] = xb[2 * NPTS + j] - pz;
    e[3] = px; e[4] = py; e[5] = pz;
#pragma unroll
    for (int i = 0; i < 6; i++) loc[i] += e[i];
    int c = 6;
#pragma unroll
    for (int i = 0; i < 6; i++)
#pragma unroll
      for (int jj = i; jj < 6; jj++) { loc[c] += e[i] * e[jj]; c++; }
  }

  __shared__ float red[256];
#pragma unroll
  for (int q = 0; q < 27; q++) {
    red[threadIdx.x] = loc[q];
    __syncthreads();
    for (int s = 128; s > 0; s >>= 1) {
      if (threadIdx.x < s) red[threadIdx.x] += red[threadIdx.x + s];
      __syncthreads();
    }
    if (threadIdx.x == 0) part[blockIdx.x * 32 + q] = red[0];
    __syncthreads();
  }
}

// ---------------------------------------------------------------------------
// Kernel 3: BN1 folded to affine:  scale1 = g1*rsqrt(var+eps), bias1 = b1-mean*scale1
// ---------------------------------------------------------------------------
__global__ void stats1_kernel(const float* __restrict__ part, int nblk,
                              const float* __restrict__ W1,
                              const float* __restrict__ g1,
                              const float* __restrict__ b1,
                              float* __restrict__ sb1) {
  __shared__ float mom[27];
  __shared__ float M2[6][6];
  __shared__ float M1[6];
  const int tid = threadIdx.x;
  if (tid < 27) {
    float s = 0.f;
    for (int i = 0; i < nblk; i++) s += part[i * 32 + tid];
    mom[tid] = s / BNKF;
  }
  __syncthreads();
  if (tid == 0) {
    int c = 0;
    for (int i = 0; i < 6; i++)
      for (int j = i; j < 6; j++) {
        float v = mom[6 + c]; M2[i][j] = v; M2[j][i] = v; c++;
      }
    for (int i = 0; i < 6; i++) M1[i] = mom[i];
  }
  __syncthreads();
  if (tid < 64) {
    float w[6];
#pragma unroll
    for (int c = 0; c < 6; c++) w[c] = W1[tid * 6 + c];
    float mean = 0.f;
#pragma unroll
    for (int c = 0; c < 6; c++) mean += w[c] * M1[c];
    float ex2 = 0.f;
#pragma unroll
    for (int i = 0; i < 6; i++)
#pragma unroll
      for (int j = 0; j < 6; j++) ex2 += w[i] * w[j] * M2[i][j];
    float var = ex2 - mean * mean;
    float sc = g1[tid] * rsqrtf(var + EPSV);
    sb1[tid] = sc;
    sb1[64 + tid] = b1[tid] - mean * sc;
  }
}

// ---------------------------------------------------------------------------
// a1 value for one (edge, channel); channel parameters passed per call.
// ---------------------------------------------------------------------------
__device__ __forceinline__ float a1_val(float e0, float e1, float e2,
                                        float px, float py, float pz,
                                        const float w[6], float sc, float bi) {
  float h = w[0] * e0 + w[1] * e1 + w[2] * e2 +
            w[3] * px + w[4] * py + w[5] * pz;
  float a = fmaf(sc, h, bi);
  return a > 0.f ? a : 0.2f * a;
}

// ---------------------------------------------------------------------------
// Kernel 4: Gram of a1 (G = A^T A, 64x64) + sum(a1), via WMMA f16->f32.
// a1 tile stored TRANSPOSED (channel-major, stride TST) so both the A^T and
// B fragments are contiguous 128-bit LDS loads.
// ---------------------------------------------------------------------------
__global__ __launch_bounds__(256) void gram_kernel(const float* __restrict__ x,
                                                   const int* __restrict__ idx,
                                                   const float* __restrict__ W1,
                                                   const float* __restrict__ sb1,
                                                   float* __restrict__ gpart) {
  __shared__ __align__(16) _Float16 tilesT[8][CH * TST];
  __shared__ float gsh[CH * CH];
  __shared__ float s1sh[CH];
  const int lane = threadIdx.x & 31, wave = threadIdx.x >> 5;

  // lane owns channel rows 2*lane and 2*lane+1
  const int c0 = 2 * lane, c1 = 2 * lane + 1;
  float wa[6], wb[6];
#pragma unroll
  for (int c = 0; c < 6; c++) {
    wa[c] = W1[c0 * 6 + c];
    wb[c] = W1[c1 * 6 + c];
  }
  const float sca = sb1[c0], bia = sb1[64 + c0];
  const float scb = sb1[c1], bib = sb1[64 + c1];

  _Float16* tile = tilesT[wave];
  // zero-pad edge columns 20..31 once (never overwritten in the loop)
  {
    const v2h z2 = {(_Float16)0.f, (_Float16)0.f};
#pragma unroll
    for (int e = KNN; e < 32; e += 2) {
      *(v2h*)(tile + c0 * TST + e) = z2;
      *(v2h*)(tile + c1 * TST + e) = z2;
    }
  }

  const v8f zv = {0.f, 0.f, 0.f, 0.f, 0.f, 0.f, 0.f, 0.f};
  v8f acc[16];
#pragma unroll
  for (int i = 0; i < 16; i++) acc[i] = zv;
  float s1a = 0.f, s1b = 0.f;

  const int gw = blockIdx.x * 8 + wave;           // 1024 waves total
#pragma unroll 1
  for (int p = gw; p < NB * NPTS; p += 1024) {
    const int b = p >> 12, n = p & (NPTS - 1);
    const float* xb = x + (size_t)b * 3 * NPTS;
    const float px = xb[n], py = xb[NPTS + n], pz = xb[2 * NPTS + n];
    const int* ip = idx + (size_t)p * KNN;

    // write a1 transposed: edge pairs -> 32-bit LDS stores
#pragma unroll 1
    for (int e = 0; e < KNN; e += 2) {
      const int j0 = ip[e], j1 = ip[e + 1];
      const float d00 = xb[j0] - px, d01 = xb[NPTS + j0] - py, d02 = xb[2 * NPTS + j0] - pz;
      const float d10 = xb[j1] - px, d11 = xb[NPTS + j1] - py, d12 = xb[2 * NPTS + j1] - pz;
      float aA0 = a1_val(d00, d01, d02, px, py, pz, wa, sca, bia);
      float aA1 = a1_val(d10, d11, d12, px, py, pz, wa, sca, bia);
      float aB0 = a1_val(d00, d01, d02, px, py, pz, wb, scb, bib);
      float aB1 = a1_val(d10, d11, d12, px, py, pz, wb, scb, bib);
      s1a += aA0 + aA1;
      s1b += aB0 + aB1;
      v2h pa; pa[0] = (_Float16)aA0; pa[1] = (_Float16)aA1;
      v2h pb; pb[0] = (_Float16)aB0; pb[1] = (_Float16)aB1;
      *(v2h*)(tile + c0 * TST + e) = pa;
      *(v2h*)(tile + c1 * TST + e) = pb;
    }

    // B fragments: B[k][n'] = a1[k][nt*16+n'] = tileT[(nt*16+n')*TST + k]
    const int nfo = lane & 15;
    const int kb2 = (lane < 16) ? 0 : 8;
    v16h Bf[4];
#pragma unroll
    for (int nt = 0; nt < 4; nt++) {
      const _Float16* br = tile + (nt * 16 + nfo) * TST + kb2;
      v8h blo = *(const v8h*)(br);
      v8h bhi = *(const v8h*)(br + 16);
      Bf[nt] = SHUF16(blo, bhi);
    }
    // A' fragments: A'[m][k] = a1[k][mt*16+m] = tileT[(mt*16+m)*TST + k]
    const int m = lane & 15;
    const int kb = (lane < 16) ? 0 : 8;
#pragma unroll
    for (int mt = 0; mt < 4; mt++) {
      const _Float16* ar = tile + (mt * 16 + m) * TST;
      v8h alo = *(const v8h*)(ar + kb);
      v8h ahi = *(const v8h*)(ar + 16 + kb);
      v16h Af = SHUF16(alo, ahi);
#pragma unroll
      for (int nt = 0; nt < 4; nt++) {
        acc[mt * 4 + nt] = __builtin_amdgcn_wmma_f32_16x16x32_f16(
            false, Af, false, Bf[nt], (short)0, acc[mt * 4 + nt], false, false);
      }
    }
  }

  // deterministic cross-wave reduction (serialized per wave)
  for (int i = threadIdx.x; i < CH * CH; i += 256) gsh[i] = 0.f;
  if (threadIdx.x < CH) s1sh[threadIdx.x] = 0.f;
  __syncthreads();
  const int nn = lane & 15;
  const int mofs = (lane < 16) ? 0 : 8;
  for (int w = 0; w < 8; w++) {
    if (wave == w) {
#pragma unroll
      for (int mt = 0; mt < 4; mt++)
#pragma unroll
        for (int nt = 0; nt < 4; nt++)
#pragma unroll
          for (int r = 0; r < 8; r++)
            gsh[(mt * 16 + r + mofs) * CH + nt * 16 + nn] += acc[mt * 4 + nt][r];
      s1sh[c0] += s1a;
      s1sh[c1] += s1b;
    }
    __syncthreads();
  }
  float* gp = gpart + (size_t)blockIdx.x * (CH * CH + CH);
  for (int i = threadIdx.x; i < CH * CH; i += 256) gp[i] = gsh[i];
  if (threadIdx.x < CH) gp[CH * CH + threadIdx.x] = s1sh[threadIdx.x];
}

// ---------------------------------------------------------------------------
// Kernel 5: BN2 folded to affine from Gram + mean(a1).
// ---------------------------------------------------------------------------
__global__ void stats2_kernel(const float* __restrict__ gpart, int nblk,
                              const float* __restrict__ W2,
                              const float* __restrict__ g2,
                              const float* __restrict__ b2,
                              float* __restrict__ sb2) {
  __shared__ float G[CH][CH];
  __shared__ float ma[CH];
  __shared__ float w2s[CH][CH];
  const int tid = threadIdx.x;          // 64 threads
  for (int c = 0; c < CH; c++) {
    float s = 0.f;
    for (int i = 0; i < nblk; i++) s += gpart[(size_t)i * (CH * CH + CH) + tid * CH + c];
    G[tid][c] = s / BNKF;
  }
  {
    float s = 0.f;
    for (int i = 0; i < nblk; i++) s += gpart[(size_t)i * (CH * CH + CH) + CH * CH + tid];
    ma[tid] = s / BNKF;
  }
  for (int c = 0; c < CH; c++) w2s[tid][c] = W2[tid * CH + c];
  __syncthreads();
  float mean = 0.f;
  for (int c = 0; c < CH; c++) mean += w2s[tid][c] * ma[c];
  float ex2 = 0.f;
  for (int c = 0; c < CH; c++) {
    const float wc = w2s[tid][c];
    float t = 0.f;
    for (int c2 = 0; c2 < CH; c2++) t += w2s[tid][c2] * G[c][c2];
    ex2 += wc * t;
  }
  const float var = ex2 - mean * mean;
  const float sc = g2[tid] * rsqrtf(var + EPSV);
  sb2[tid] = sc;
  sb2[64 + tid] = b2[tid] - mean * sc;
}

// ---------------------------------------------------------------------------
// Kernel 6: h2 = a1 @ W2^T via WMMA, BN2 affine, leaky-relu, max over K, store.
// a1 tile row-major (edge-major): A and B fragments are contiguous 128-bit loads.
// ---------------------------------------------------------------------------
__global__ __launch_bounds__(256) void conv2_kernel(const float* __restrict__ x,
                                                    const int* __restrict__ idx,
                                                    const float* __restrict__ W1,
                                                    const float* __restrict__ sb1,
                                                    const float* __restrict__ W2,
                                                    const float* __restrict__ sb2,
                                                    float* __restrict__ out) {
  __shared__ __align__(16) _Float16 tiles[8][32 * CH];
  __shared__ __align__(16) _Float16 w2h[CH * CH];
  const int lane = threadIdx.x & 31, wave = threadIdx.x >> 5;
  for (int i = threadIdx.x; i < CH * CH; i += 256) w2h[i] = (_Float16)W2[i];
  __syncthreads();

  // lane owns channel pair (2*lane, 2*lane+1) when producing a1
  const int c0 = 2 * lane, c1 = 2 * lane + 1;
  float wa[6], wb[6];
#pragma unroll
  for (int c = 0; c < 6; c++) {
    wa[c] = W1[c0 * 6 + c];
    wb[c] = W1[c1 * 6 + c];
  }
  const float sca = sb1[c0], bia = sb1[64 + c0];
  const float scb = sb1[c1], bib = sb1[64 + c1];

  _Float16* tile = tiles[wave];
  // zero-pad edge rows 20..31 once (never overwritten in the loop)
  {
    const v2h z2 = {(_Float16)0.f, (_Float16)0.f};
#pragma unroll
    for (int e = KNN; e < 32; e++) *(v2h*)(tile + e * CH + c0) = z2;
  }

  // W2^T fragments: B[k][n'] = W2[nt*16+n'][ks*32+k] -> contiguous k run
  const int nfo = lane & 15;
  const int kb2 = (lane < 16) ? 0 : 8;
  v16h Bf[2][4];
#pragma unroll
  for (int ks = 0; ks < 2; ks++)
#pragma unroll
    for (int nt = 0; nt < 4; nt++) {
      const _Float16* br = w2h + (nt * 16 + nfo) * CH + ks * 32 + kb2;
      v8h blo = *(const v8h*)(br);
      v8h bhi = *(const v8h*)(br + 16);
      Bf[ks][nt] = SHUF16(blo, bhi);
    }
  float s2[4], bi2[4];
#pragma unroll
  for (int nt = 0; nt < 4; nt++) {
    s2[nt] = sb2[nt * 16 + nfo];
    bi2[nt] = sb2[64 + nt * 16 + nfo];
  }

  const v8f zv = {0.f, 0.f, 0.f, 0.f, 0.f, 0.f, 0.f, 0.f};
  const int gw = blockIdx.x * 8 + wave;           // 4096 waves total
#pragma unroll 1
  for (int p = gw; p < NB * NPTS; p += 4096) {
    const int b = p >> 12, n = p & (NPTS - 1);
    const float* xb = x + (size_t)b * 3 * NPTS;
    const float px = xb[n], py = xb[NPTS + n], pz = xb[2 * NPTS + n];
    const int* ip = idx + (size_t)p * KNN;

    // write a1 row-major: channel pair -> 32-bit LDS store per edge
#pragma unroll 1
    for (int e = 0; e < KNN; e++) {
      const int j = ip[e];
      const float e0 = xb[j] - px, e1 = xb[NPTS + j] - py, e2 = xb[2 * NPTS + j] - pz;
      float a0 = a1_val(e0, e1, e2, px, py, pz, wa, sca, bia);
      float a1v = a1_val(e0, e1, e2, px, py, pz, wb, scb, bib);
      v2h pr; pr[0] = (_Float16)a0; pr[1] = (_Float16)a1v;
      *(v2h*)(tile + e * CH + c0) = pr;
    }

    float best[4] = {-1e30f, -1e30f, -1e30f, -1e30f};
    const int m = lane & 15;
    const int kb = (lane < 16) ? 0 : 8;
#pragma unroll
    for (int mt = 0; mt < 2; mt++) {
      // A fragments: A[m][k] = a1[mt*16+m][ks*32+k] -> two contiguous 8-runs
      v16h Af[2];
#pragma unroll
      for (int ks = 0; ks < 2; ks++) {
        const _Float16* ar = tile + (mt * 16 + m) * CH + ks * 32;
        v8h alo = *(const v8h*)(ar + kb);
        v8h ahi = *(const v8h*)(ar + 16 + kb);
        Af[ks] = SHUF16(alo, ahi);
      }
#pragma unroll
      for (int nt = 0; nt < 4; nt++) {
        v8f accv = zv;
#pragma unroll
        for (int ks = 0; ks < 2; ks++)
          accv = __builtin_amdgcn_wmma_f32_16x16x32_f16(
              false, Af[ks], false, Bf[ks][nt], (short)0, accv, false, false);
        float pm = -1e30f;
        const int rofs = (lane < 16) ? 0 : 8;
#pragma unroll
        for (int r = 0; r < 8; r++) {
          const int row = mt * 16 + r + rofs;
          float v = fmaf(s2[nt], accv[r], bi2[nt]);
          v = v > 0.f ? v : 0.2f * v;
          pm = (row < KNN) ? fmaxf(pm, v) : pm;
        }
        pm = fmaxf(pm, __shfl_xor(pm, 16, 32));
        best[nt] = fmaxf(best[nt], pm);
      }
    }
    if (lane < 16) {
#pragma unroll
      for (int nt = 0; nt < 4; nt++)
        out[((size_t)b * CH + nt * 16 + lane) * NPTS + n] = best[nt];
    }
  }
}

// ---------------------------------------------------------------------------
extern "C" void kernel_launch(void* const* d_in, const int* in_sizes, int n_in,
                              void* d_out, int out_size, void* d_ws, size_t ws_size,
                              hipStream_t stream) {
  (void)in_sizes; (void)n_in; (void)out_size; (void)ws_size;
  const float* x  = (const float*)d_in[0];
  const float* W1 = (const float*)d_in[1];
  const float* g1 = (const float*)d_in[2];
  const float* b1 = (const float*)d_in[3];
  const float* W2 = (const float*)d_in[4];
  const float* g2 = (const float*)d_in[5];
  const float* b2 = (const float*)d_in[6];
  float* out = (float*)d_out;

  char* ws = (char*)d_ws;
  int*   idx   = (int*)ws;                    // 8*4096*20*4       = 2,621,440 B
  float* mpart = (float*)(ws + 2621440);      // 128 blocks * 32f  =    16,384 B
  float* sb1   = (float*)(ws + 2637824);      // 128f              =       512 B
  float* gpart = (float*)(ws + 2638336);      // 128 * 4160f       = 2,129,920 B
  float* sb2   = (float*)(ws + 4768256);      // 128f              =       512 B

  knn_kernel<<<dim3(16, 8), 256, 0, stream>>>(x, idx);
  moments_kernel<<<128, 256, 0, stream>>>(x, idx, mpart);
  stats1_kernel<<<1, 64, 0, stream>>>(mpart, 128, W1, g1, b1, sb1);
  gram_kernel<<<128, 256, 0, stream>>>(x, idx, W1, sb1, gpart);
  stats2_kernel<<<1, 64, 0, stream>>>(gpart, 128, W2, g2, b2, sb2);
  conv2_kernel<<<512, 256, 0, stream>>>(x, idx, W1, sb1, W2, sb2, out);
}
